// Experts_23210003268115
// MI455X (gfx1250) — matrix operand order repaired and checked by reading, not compiled, not run on hardware
//
#include <hip/hip_runtime.h>

#define N_TOK    8192
#define C_DIM    1024
#define H_DIM    3072
#define E_NUM    8
#define BM       128
#define BN       64
#define KT       32
#define MAXROWS  17408          // >= sum of 128-aligned per-expert counts
#define MAXTILES 136            // MAXROWS / BM
#define ASTRIDE  48             // halves; 96 B row, 16 B aligned
#define BTSTRIDE 48             // halves; 96 B row, 16 B aligned

typedef __attribute__((ext_vector_type(16))) __bf16          v16bf;
typedef __attribute__((ext_vector_type(16))) unsigned short  us16;
typedef __attribute__((ext_vector_type(8)))  float           v8f;

union Frag16 { uint4 q[2]; us16 v; };

__device__ __forceinline__ unsigned int bfpack2(float a, float b) {
  unsigned ua = __builtin_bit_cast(unsigned int, a);
  unsigned ub = __builtin_bit_cast(unsigned int, b);
  ua = (ua + 0x7FFFu + ((ua >> 16) & 1u)) >> 16;   // RNE
  ub = (ub + 0x7FFFu + ((ub >> 16) & 1u)) >> 16;
  return (ub << 16) | (ua & 0xFFFFu);
}
__device__ __forceinline__ unsigned short bf16r(float a) {
  unsigned ua = __builtin_bit_cast(unsigned int, a);
  return (unsigned short)((ua + 0x7FFFu + ((ua >> 16) & 1u)) >> 16);
}

// ---------------------------------------------------------------- utilities
__global__ void zero_u32_kernel(unsigned int* __restrict__ p, int n) {
  int i = blockIdx.x * blockDim.x + threadIdx.x;
  if (i < n) p[i] = 0u;
}

// fp32 W[E][K][N] -> bf16 Wt[E][N][K] via 32x32 LDS tile transpose
__global__ __launch_bounds__(256) void transpose_convert_kernel(
    const float* __restrict__ W, unsigned short* __restrict__ Wt,
    int K, int N) {
  __shared__ unsigned short T[32][36];
  const int k0 = blockIdx.x * 32, n0 = blockIdx.y * 32;
  const float* We = W + (size_t)blockIdx.z * K * N;
  unsigned short* Wte = Wt + (size_t)blockIdx.z * K * N;
  {
    int r  = threadIdx.x >> 3;          // k within tile
    int c4 = (threadIdx.x & 7) * 4;     // n within tile
    float4 v = *(const float4*)(We + (size_t)(k0 + r) * N + n0 + c4);
    T[c4 + 0][r] = bf16r(v.x);
    T[c4 + 1][r] = bf16r(v.y);
    T[c4 + 2][r] = bf16r(v.z);
    T[c4 + 3][r] = bf16r(v.w);
  }
  __syncthreads();
  {
    int n  = threadIdx.x >> 3;          // row of transposed tile
    int k4 = (threadIdx.x & 7) * 4;
    uint2 p = *(const uint2*)&T[n][k4];
    *(uint2*)(Wte + (size_t)(n0 + n) * K + k0 + k4) = p;
  }
}

// ---------------------------------------------------------------- router
__global__ __launch_bounds__(256) void router_kernel(
    const float* __restrict__ x, const float* __restrict__ Wr,
    float* __restrict__ probs, int2* __restrict__ tidx,
    float2* __restrict__ tw, int* __restrict__ counts) {
  const int wave = threadIdx.x >> 5, lane = threadIdx.x & 31;
  const int n = blockIdx.x * 8 + wave;
  if (n >= N_TOK) return;
  float acc[E_NUM];
#pragma unroll
  for (int e = 0; e < E_NUM; ++e) acc[e] = 0.f;
  for (int c = lane; c < C_DIM; c += 32) {
    float xv = x[(size_t)n * C_DIM + c];
    const float4* wr = (const float4*)(Wr + (size_t)c * E_NUM);
    float4 w0 = wr[0], w1 = wr[1];
    acc[0] += xv * w0.x; acc[1] += xv * w0.y; acc[2] += xv * w0.z; acc[3] += xv * w0.w;
    acc[4] += xv * w1.x; acc[5] += xv * w1.y; acc[6] += xv * w1.z; acc[7] += xv * w1.w;
  }
#pragma unroll
  for (int off = 16; off > 0; off >>= 1)
#pragma unroll
    for (int e = 0; e < E_NUM; ++e) acc[e] += __shfl_xor(acc[e], off, 32);
  if (lane == 0) {
    float m = acc[0];
#pragma unroll
    for (int e = 1; e < E_NUM; ++e) m = fmaxf(m, acc[e]);
    float p[E_NUM], s = 0.f;
#pragma unroll
    for (int e = 0; e < E_NUM; ++e) { p[e] = __expf(acc[e] - m); s += p[e]; }
    float inv = __builtin_amdgcn_rcpf(s);
#pragma unroll
    for (int e = 0; e < E_NUM; ++e) { p[e] *= inv; probs[(size_t)n * E_NUM + e] = p[e]; }
    int i0 = 0; float v0 = p[0];
#pragma unroll
    for (int e = 1; e < E_NUM; ++e) if (p[e] > v0) { v0 = p[e]; i0 = e; }
    int i1 = (i0 == 0) ? 1 : 0; float v1 = p[i1];
#pragma unroll
    for (int e = 0; e < E_NUM; ++e)
      if (e != i0 && p[e] > v1) { v1 = p[e]; i1 = e; }
    float r = 1.f / (v0 + v1);
    tidx[n] = make_int2(i0, i1);
    tw[n]   = make_float2(v0 * r, v1 * r);
    atomicAdd(&counts[i0], 1);
    atomicAdd(&counts[i1], 1);
  }
}

__global__ void bases_kernel(const int* __restrict__ counts,
                             int* __restrict__ base, int* __restrict__ cursor) {
  if (threadIdx.x == 0 && blockIdx.x == 0) {
    int b = 0;
    for (int e = 0; e < E_NUM; ++e) {
      base[e] = b;
      b += ((counts[e] + BM - 1) / BM) * BM;
      cursor[e] = 0;
    }
    base[E_NUM] = b;   // padded total rows (multiple of 128, <= MAXROWS)
  }
}

__global__ void assign_kernel(const int2* __restrict__ tidx,
                              const int* __restrict__ base,
                              int* __restrict__ cursor,
                              int* __restrict__ slot_of) {
  int n = blockIdx.x * blockDim.x + threadIdx.x;
  if (n >= N_TOK) return;
  int2 e = tidx[n];
  slot_of[2 * n]     = base[e.x] + atomicAdd(&cursor[e.x], 1);
  slot_of[2 * n + 1] = base[e.y] + atomicAdd(&cursor[e.y], 1);
}

__global__ __launch_bounds__(256) void gather_kernel(
    const float* __restrict__ x, const int* __restrict__ slot_of,
    unsigned short* __restrict__ xg) {
  const int n = blockIdx.x, t = threadIdx.x;     // 256 threads, 4 floats each
  const int s0 = slot_of[2 * n], s1 = slot_of[2 * n + 1];
  float4 v = ((const float4*)(x + (size_t)n * C_DIM))[t];
  uint2 p; p.x = bfpack2(v.x, v.y); p.y = bfpack2(v.z, v.w);
  ((uint2*)(xg + (size_t)s0 * C_DIM))[t] = p;
  ((uint2*)(xg + (size_t)s1 * C_DIM))[t] = p;
}

// ---------------------------------------------------------------- WMMA GEMM
// Out[row, col] = act( sum_k A[row,k] * Wt[e][col,k] ), rows grouped by expert.
// A: bf16 [rows, KDIM] row-major. Wt: bf16 [E][NDIM][KDIM] (pre-transposed).
// Block tile 128x64, 8 waves (4x2), each wave 32x32 via 2x2 WMMA bf16 k32.
template <int KDIM, int NDIM, bool SILU, bool OUTBF16>
__global__ __launch_bounds__(256) void moe_gemm_kernel(
    const unsigned short* __restrict__ A,
    const unsigned short* __restrict__ Wt,
    void* __restrict__ OutP,
    const int* __restrict__ base) {
  __shared__ unsigned short As[BM][ASTRIDE];    // 12 KB
  __shared__ unsigned short Bt[BN][BTSTRIDE];   // 6 KB   (Bt[n][k])

  const int row0 = blockIdx.x * BM;
  if (row0 >= base[E_NUM]) return;
  int e = 0;
#pragma unroll
  for (int i = 1; i < E_NUM; ++i) if (row0 >= base[i]) e = i;
  const unsigned short* We = Wt + (size_t)e * KDIM * NDIM;
  const int col0 = blockIdx.y * BN;

  const int tid = threadIdx.x;
  const int lane = tid & 31, wave = tid >> 5;
  const int wm = wave >> 1, wn = wave & 1;          // 4x2 wave grid
  const int m = lane & 15, hsel = lane >> 4;

  // staging addresses (invariant parts)
  const int ar = tid >> 1;                    // A: 2 threads/row, 16 halves each
  const int aseg = (tid & 1) * 16;
  const int bn = tid >> 2;                    // B: 4 threads/row, 8 halves each
  const int bk = (tid & 3) * 8;
  const unsigned short* asrc = A  + (size_t)(row0 + ar) * KDIM + aseg;
  const unsigned short* bsrc = We + (size_t)(col0 + bn) * KDIM + bk;

  v8f acc[2][2] = {};

  for (int k0 = 0; k0 < KDIM; k0 += KT) {
    uint4 av0 = *(const uint4*)(asrc + k0);
    uint4 av1 = *(const uint4*)(asrc + k0 + 8);
    uint4 bv  = *(const uint4*)(bsrc + k0);
    if (k0 + KT < KDIM) {
      __builtin_prefetch(asrc + k0 + KT, 0, 1);
      __builtin_prefetch(bsrc + k0 + KT, 0, 1);
    }
    __syncthreads();
    *(uint4*)&As[ar][aseg]     = av0;
    *(uint4*)&As[ar][aseg + 8] = av1;
    *(uint4*)&Bt[bn][bk]       = bv;
    __syncthreads();

    // fragments: both A and B are 2x b128 per the gfx1250 wave32 layout
    Frag16 au[2], bu[2];
#pragma unroll
    for (int mi = 0; mi < 2; ++mi) {
      const unsigned short* ap = &As[wm * 32 + mi * 16 + m][hsel * 8];
      au[mi].q[0] = *(const uint4*)ap;          // k = hsel*8 + 0..7
      au[mi].q[1] = *(const uint4*)(ap + 16);   // k = 16 + hsel*8 + 0..7
    }
#pragma unroll
    for (int ni = 0; ni < 2; ++ni) {
      const unsigned short* bp = &Bt[wn * 32 + ni * 16 + m][hsel * 8];
      bu[ni].q[0] = *(const uint4*)bp;          // k = hsel*8 + 0..7
      bu[ni].q[1] = *(const uint4*)(bp + 16);   // k = 16 + hsel*8 + 0..7
    }
#pragma unroll
    for (int mi = 0; mi < 2; ++mi) {
      v16bf a = __builtin_bit_cast(v16bf, au[mi].v);
#pragma unroll
      for (int ni = 0; ni < 2; ++ni) {
        v16bf b = __builtin_bit_cast(v16bf, bu[ni].v);
        acc[mi][ni] = __builtin_amdgcn_wmma_f32_16x16x32_bf16(
            false, a, false, b, (short)0, acc[mi][ni], false, false);
      }
    }
  }

  // epilogue: D layout -> M = r + 8*(lane>>4), N = lane&15
#pragma unroll
  for (int mi = 0; mi < 2; ++mi)
#pragma unroll
    for (int ni = 0; ni < 2; ++ni)
#pragma unroll
      for (int r = 0; r < 8; ++r) {
        int grow = row0 + wm * 32 + mi * 16 + r + 8 * hsel;
        int gcol = col0 + wn * 32 + ni * 16 + m;
        float v = acc[mi][ni][r];
        if (SILU) v = v * __builtin_amdgcn_rcpf(1.f + __expf(-v));
        if (OUTBF16)
          ((unsigned short*)OutP)[(size_t)grow * NDIM + gcol] = bf16r(v);
        else
          ((float*)OutP)[(size_t)grow * NDIM + gcol] = v;
      }
}

// ---------------------------------------------------------------- combine
// one block per token: out[n,:] = w0*y[slot0,:] + w1*y[slot1,:]
__global__ __launch_bounds__(256) void combine_kernel(
    const float* __restrict__ y, const int* __restrict__ slot_of,
    const float2* __restrict__ tw, float* __restrict__ out) {
  const int n = blockIdx.x, c4 = threadIdx.x;   // 256 float4 = 1024 floats
  float2 w = tw[n];
  float4 a = ((const float4*)(y + (size_t)slot_of[2 * n]     * C_DIM))[c4];
  float4 b = ((const float4*)(y + (size_t)slot_of[2 * n + 1] * C_DIM))[c4];
  float4 o;
  o.x = w.x * a.x + w.y * b.x;
  o.y = w.x * a.y + w.y * b.y;
  o.z = w.x * a.z + w.y * b.z;
  o.w = w.x * a.w + w.y * b.w;
  ((float4*)(out + (size_t)n * C_DIM))[c4] = o;
}

// aux loss, deterministic fixed-order tree reduction
__global__ __launch_bounds__(256) void aux_kernel(
    const float* __restrict__ probs, const int* __restrict__ counts,
    float* __restrict__ out) {
  __shared__ float red[256];
  const int tid = threadIdx.x;
  float s[E_NUM];
#pragma unroll
  for (int e = 0; e < E_NUM; ++e) s[e] = 0.f;
  for (int t = tid; t < N_TOK; t += 256) {
    const float* pr = probs + (size_t)t * E_NUM;
#pragma unroll
    for (int e = 0; e < E_NUM; ++e) s[e] += pr[e];
  }
  float aux = 0.f;
  for (int e = 0; e < E_NUM; ++e) {
    red[tid] = s[e];
    __syncthreads();
    for (int st = 128; st > 0; st >>= 1) {
      if (tid < st) red[tid] += red[tid + st];
      __syncthreads();
    }
    if (tid == 0) {
      float pi = red[0] / (float)N_TOK;
      float fi = (float)counts[e] / (float)N_TOK;
      aux += fi * pi;
    }
    __syncthreads();
  }
  if (tid == 0) out[(size_t)N_TOK * C_DIM] = 0.01f * (float)E_NUM * aux;
}

// ---------------------------------------------------------------- launcher
extern "C" void kernel_launch(void* const* d_in, const int* in_sizes, int n_in,
                              void* d_out, int out_size, void* d_ws, size_t ws_size,
                              hipStream_t stream) {
  (void)in_sizes; (void)n_in; (void)out_size; (void)ws_size;
  const float* x  = (const float*)d_in[0];
  const float* W1 = (const float*)d_in[1];
  const float* W2 = (const float*)d_in[2];
  const float* Wr = (const float*)d_in[3];
  float* out = (float*)d_out;

  char* ws = (char*)d_ws;
  size_t off = 0;
  auto take = [&](size_t bytes) -> char* {
    char* p = ws + off;
    off = (off + bytes + 255) & ~(size_t)255;
    return p;
  };
  const size_t WBYTES = (size_t)E_NUM * C_DIM * H_DIM * 2;        // 50.3 MB bf16
  const size_t YBYTES = (size_t)MAXROWS * C_DIM * 4;              // 71.3 MB fp32

  float*          probs   = (float*)take((size_t)N_TOK * E_NUM * 4);
  int2*           tidx    = (int2*)take((size_t)N_TOK * 8);
  float2*         tw      = (float2*)take((size_t)N_TOK * 8);
  int*            counts  = (int*)take(64);
  int*            base    = (int*)take(64);
  int*            cursor  = (int*)take(64);
  int*            slot_of = (int*)take((size_t)N_TOK * 2 * 4);
  unsigned short* xg      = (unsigned short*)take((size_t)MAXROWS * C_DIM * 2);
  unsigned short* h       = (unsigned short*)take((size_t)MAXROWS * H_DIM * 2);
  char*           regA    = take(YBYTES > WBYTES ? YBYTES : WBYTES);
  unsigned short* W1t     = (unsigned short*)regA;   // [E][H][C], dead after GEMM1
  float*          y       = (float*)regA;            // [MAXROWS][C], aliases W1t
  unsigned short* W2t     = (unsigned short*)take(WBYTES);        // [E][C][H]

  zero_u32_kernel<<<1, 64, 0, stream>>>((unsigned int*)counts, 16);
  router_kernel<<<N_TOK / 8, 256, 0, stream>>>(x, Wr, probs, tidx, tw, counts);
  bases_kernel<<<1, 1, 0, stream>>>(counts, base, cursor);
  assign_kernel<<<(N_TOK + 255) / 256, 256, 0, stream>>>(tidx, base, cursor, slot_of);
  {
    int nz = MAXROWS * C_DIM / 2;  // xg as uints
    zero_u32_kernel<<<(nz + 255) / 256, 256, 0, stream>>>((unsigned int*)xg, nz);
  }
  gather_kernel<<<N_TOK, 256, 0, stream>>>(x, slot_of, xg);

  // pre-transpose weights to bf16 [E][N][K]
  transpose_convert_kernel<<<dim3(C_DIM / 32, H_DIM / 32, E_NUM), 256, 0, stream>>>(
      W1, W1t, C_DIM, H_DIM);
  transpose_convert_kernel<<<dim3(H_DIM / 32, C_DIM / 32, E_NUM), 256, 0, stream>>>(
      W2, W2t, H_DIM, C_DIM);

  moe_gemm_kernel<C_DIM, H_DIM, true,  true ><<<dim3(MAXTILES, H_DIM / BN), 256, 0, stream>>>(
      xg, W1t, (void*)h, base);
  moe_gemm_kernel<H_DIM, C_DIM, false, false><<<dim3(MAXTILES, C_DIM / BN), 256, 0, stream>>>(
      h, W2t, (void*)y, base);

  combine_kernel<<<N_TOK, 256, 0, stream>>>(y, slot_of, tw, out);
  aux_kernel<<<1, 256, 0, stream>>>(probs, counts, out);
}
